// GNN_MIL_Attn_14980845929101
// MI455X (gfx1250) — matrix operand order repaired and checked by reading, not compiled, not run on hardware
//
#include <hip/hip_runtime.h>

#define CDIV(a,b) (((a)+(b)-1)/(b))

typedef __attribute__((ext_vector_type(16))) _Float16 v16h;
typedef __attribute__((ext_vector_type(8)))  _Float16 v8h;
typedef __attribute__((ext_vector_type(8)))  float    v8f;

// ---- problem constants (from reference) ----
static const int NN   = 20000;       // nodes
static const int EE   = 160000;      // edges (before self loops)
static const int ET   = EE + NN;     // edges + self loops
static const int IND  = 512;         // in_dim
static const int HIDc = 128;         // hidden channels
static const int HDS  = 8;           // heads (conv1)
static const int F1   = HDS * HIDc;  // 1024
static const int ATTD = 128;         // MIL attn dim
static const int NCLS = 7;
static const int PP   = 500;         // patients

// ---------------- device helpers ----------------
__device__ __forceinline__ unsigned f2ord(float f) {
  unsigned u = __float_as_uint(f);
  return (u & 0x80000000u) ? ~u : (u | 0x80000000u);
}
__device__ __forceinline__ float ord2f(unsigned u) {
  return __uint_as_float((u & 0x80000000u) ? (u ^ 0x80000000u) : ~u);
}

// A fragment (16x32 f16, MxK): lane<16 -> row, K={k..k+15}; lane>=16 shifted by 16
__device__ __forceinline__ v16h load_a_frag(const _Float16* p) {
  v8h lo = *reinterpret_cast<const v8h*>(p);
  v8h hi = *reinterpret_cast<const v8h*>(p + 8);
  return __builtin_shufflevector(lo, hi, 0,1,2,3,4,5,6,7,8,9,10,11,12,13,14,15);
}

// ---------------- WMMA GEMM: double-buffered async-LDS staged B ----------------
// C[M,N] (f32) = A[M,K] (f16 row-major) * B, with B given as BT[N,K] (f16 row-major).
// Workgroup = 8 waves covering 128(M) x 64(N). K-slab = 64:
//   - slab (64 cols x 64 K halves = 8KB) staged via 2x global_load_async_to_lds_b128/thread
//   - double-buffered: next slab's DMA issued before waiting on current
//     (async loads complete in order -> s_wait_asynccnt 2 releases current slab
//      while the 2 prefetch chunks stay in flight)
//   - each wave: 2 x (A frag from global + 4 B frags from LDS + 4 WMMA) per slab
__global__ void __launch_bounds__(256)
wmma_gemm_lds(const _Float16* __restrict__ A, const _Float16* __restrict__ BT,
              float* __restrict__ C, int M, int N, int K) {
  __shared__ __align__(32) _Float16 shB[2][64 * 64];   // 2 x 8KB, [col][k]

  int tid    = threadIdx.x;
  int waveId = tid >> 5;
  int lane   = tid & 31;
  int nBlk   = N >> 6;
  int wgN    = blockIdx.x % nBlk;
  int wgM    = blockIdx.x / nBlk;
  int tn     = wgN << 6;
  int tm     = wgM * 128 + (waveId << 4);
  // wave-uniform predicate, scalarized so the compiler branches (EXEC untouched around WMMA)
  bool active = __builtin_amdgcn_readfirstlane((int)(tm + 16 <= M)) != 0;

  int half_sel = lane >> 4;
  int l15      = lane & 15;

  // staging map: 512 x 16B chunks per slab, 2 per thread (col, col+32 at same octant)
  int scol = tid >> 3, soct = tid & 7;
  const _Float16* bsrc0 = BT + (size_t)(tn + scol) * K + soct * 8;
  const _Float16* bsrc1 = BT + (size_t)(tn + scol + 32) * K + soct * 8;
  unsigned lds0[2], lds1[2];
  lds0[0] = (unsigned)(unsigned long long)(&shB[0][scol * 64 + soct * 8]);
  lds1[0] = lds0[0] + 32 * 64 * 2;   // col+32 -> +4096 bytes
  lds0[1] = (unsigned)(unsigned long long)(&shB[1][scol * 64 + soct * 8]);
  lds1[1] = lds0[1] + 32 * 64 * 2;

  const _Float16* Ap = A + (size_t)(tm + l15) * K + (half_sel << 4);
  unsigned fb = l15 * 64 + half_sel * 16;   // fragment element offset within a buffer

  auto stage = [&](int k, int buf) {
    unsigned long long g0 = (unsigned long long)(bsrc0 + k);
    unsigned long long g1 = (unsigned long long)(bsrc1 + k);
    asm volatile("global_load_async_to_lds_b128 %0, %1, off"
                 :: "v"(lds0[buf]), "v"(g0) : "memory");
    asm volatile("global_load_async_to_lds_b128 %0, %1, off"
                 :: "v"(lds1[buf]), "v"(g1) : "memory");
  };

  v8f a0 = {}, a1 = {}, a2 = {}, a3 = {};
  stage(0, 0);
  int buf = 0;
  for (int k = 0; k < K; k += 64, buf ^= 1) {
    bool more = (k + 64) < K;                 // scalar (k, K uniform)
    if (more) {
      stage(k + 64, buf ^ 1);                 // prefetch next slab into other buffer
      asm volatile("s_wait_asynccnt 2" ::: "memory");   // current slab resident
    } else {
      asm volatile("s_wait_asynccnt 0" ::: "memory");
    }
    __syncthreads();
    if (active) {
      const _Float16* bb = &shB[buf][0];
      // K sub-step 0
      v16h af = load_a_frag(Ap + k);
      v16h b0 = *reinterpret_cast<const v16h*>(bb + fb);
      v16h b1 = *reinterpret_cast<const v16h*>(bb + fb + 16 * 64);
      v16h b2 = *reinterpret_cast<const v16h*>(bb + fb + 32 * 64);
      v16h b3 = *reinterpret_cast<const v16h*>(bb + fb + 48 * 64);
      a0 = __builtin_amdgcn_wmma_f32_16x16x32_f16(false, af, false, b0, (short)0, a0, false, false);
      a1 = __builtin_amdgcn_wmma_f32_16x16x32_f16(false, af, false, b1, (short)0, a1, false, false);
      a2 = __builtin_amdgcn_wmma_f32_16x16x32_f16(false, af, false, b2, (short)0, a2, false, false);
      a3 = __builtin_amdgcn_wmma_f32_16x16x32_f16(false, af, false, b3, (short)0, a3, false, false);
      // K sub-step 1 (+32 halves within the slab)
      af = load_a_frag(Ap + k + 32);
      b0 = *reinterpret_cast<const v16h*>(bb + fb + 32);
      b1 = *reinterpret_cast<const v16h*>(bb + fb + 32 + 16 * 64);
      b2 = *reinterpret_cast<const v16h*>(bb + fb + 32 + 32 * 64);
      b3 = *reinterpret_cast<const v16h*>(bb + fb + 32 + 48 * 64);
      a0 = __builtin_amdgcn_wmma_f32_16x16x32_f16(false, af, false, b0, (short)0, a0, false, false);
      a1 = __builtin_amdgcn_wmma_f32_16x16x32_f16(false, af, false, b1, (short)0, a1, false, false);
      a2 = __builtin_amdgcn_wmma_f32_16x16x32_f16(false, af, false, b2, (short)0, a2, false, false);
      a3 = __builtin_amdgcn_wmma_f32_16x16x32_f16(false, af, false, b3, (short)0, a3, false, false);
    }
    __syncthreads();                          // protect buf^1 before next prefetch
  }
  if (active) {
    // C/D layout: VGPR r, lane l -> row tm + r + (l>=16)*8, col tn + (l&15)
    int rbase = tm + (half_sel << 3);
    for (int r = 0; r < 8; ++r) {
      float* crow = C + (size_t)(rbase + r) * N + tn + l15;
      crow[0]  = a0[r];
      crow[16] = a1[r];
      crow[32] = a2[r];
      crow[48] = a3[r];
    }
  }
}

// ---------------- elementwise / prep kernels ----------------
__global__ void fill_f32_k(float* p, float v, int n) {
  int i = blockIdx.x * blockDim.x + threadIdx.x;
  if (i < n) p[i] = v;
}
__global__ void cvt_f16_k(const float* __restrict__ in, _Float16* __restrict__ out, int n) {
  int i = blockIdx.x * blockDim.x + threadIdx.x;
  if (i < n) out[i] = (_Float16)in[i];
}
// in[R,C] f32 -> out[C,R] f16
__global__ void transpose_f16_k(const float* __restrict__ in, _Float16* __restrict__ out, int R, int C) {
  int i = blockIdx.x * blockDim.x + threadIdx.x;
  if (i >= R * C) return;
  int r = i / C, c = i - r * C;
  out[(size_t)c * R + r] = (_Float16)in[i];
}
// a[i] = sum_c XL[i*C + c] * att[(i%H)*C + c]   (i = n*H + h)
__global__ void attn_dot_k(const float* __restrict__ XL, const float* __restrict__ att,
                           float* __restrict__ out, int nH, int H, int C) {
  int i = blockIdx.x * blockDim.x + threadIdx.x;
  if (i >= nH) return;
  const float* xr = XL + (size_t)i * C;
  const float* ar = att + (size_t)(i % H) * C;
  float v = 0.f;
  for (int c = 0; c < C; ++c) v += xr[c] * ar[c];
  out[i] = v;
}

// ---------------- edge (segment softmax + aggregate) kernels ----------------
__global__ void edge_logits_k(const int* __restrict__ ei, const float* __restrict__ as,
                              const float* __restrict__ ad, float* __restrict__ ebuf,
                              unsigned* __restrict__ mbuf, int E, int Etot, int H) {
  int t = blockIdx.x * blockDim.x + threadIdx.x;
  if (t >= Etot * H) return;
  int e = t / H, h = t - e * H;
  int s = (e < E) ? ei[e]     : (e - E);
  int d = (e < E) ? ei[E + e] : (e - E);
  float v = as[s * H + h] + ad[d * H + h];
  v = (v > 0.f) ? v : 0.2f * v;               // leaky_relu, slope 0.2
  ebuf[t] = v;
  atomicMax(&mbuf[d * H + h], f2ord(v));
}
__global__ void edge_expsum_k(const int* __restrict__ ei, const unsigned* __restrict__ mbuf,
                              float* __restrict__ ebuf, float* __restrict__ sbuf,
                              int E, int Etot, int H) {
  int t = blockIdx.x * blockDim.x + threadIdx.x;
  if (t >= Etot * H) return;
  int e = t / H, h = t - e * H;
  int d = (e < E) ? ei[E + e] : (e - E);
  float ex = __expf(ebuf[t] - ord2f(mbuf[d * H + h]));
  ebuf[t] = ex;
  atomicAdd(&sbuf[d * H + h], ex);
}
__global__ void edge_aggr_k(const int* __restrict__ ei, const float* __restrict__ XL,
                            const float* __restrict__ ebuf, const float* __restrict__ sbuf,
                            float* __restrict__ out, int E, int Etot, int H, int C) {
  int t = blockIdx.x * blockDim.x + threadIdx.x;
  if (t >= Etot * H) return;
  int e = t / H, h = t - e * H;
  int s = (e < E) ? ei[e]     : (e - E);
  int d = (e < E) ? ei[E + e] : (e - E);
  float alpha = ebuf[t] / (sbuf[d * H + h] + 1e-16f);
  const float* xs = XL + ((size_t)s * H + h) * C;
  float* od = out + ((size_t)d * H + h) * C;
  for (int c = 0; c < C; ++c) atomicAdd(&od[c], xs[c] * alpha);
}

// ---------------- epilogues ----------------
__global__ void bias_elu_f16_k(const float* __restrict__ acc, const float* __restrict__ b,
                               _Float16* __restrict__ outh, int n, int ncols) {
  int i = blockIdx.x * blockDim.x + threadIdx.x;
  if (i >= n) return;
  float v = acc[i] + b[i % ncols];
  v = (v > 0.f) ? v : (__expf(v) - 1.f);      // elu
  outh[i] = (_Float16)v;
}
__global__ void bias_elu_both_k(const float* __restrict__ acc, const float* __restrict__ b,
                                float* __restrict__ outf, _Float16* __restrict__ outh,
                                int n, int ncols) {
  int i = blockIdx.x * blockDim.x + threadIdx.x;
  if (i >= n) return;
  float v = acc[i] + b[i % ncols];
  v = (v > 0.f) ? v : (__expf(v) - 1.f);
  outf[i] = v;
  outh[i] = (_Float16)v;
}
__global__ void tanh_bias_k(float* __restrict__ acc, const float* __restrict__ b, int n, int ncols) {
  int i = blockIdx.x * blockDim.x + threadIdx.x;
  if (i >= n) return;
  acc[i] = tanhf(acc[i] + b[i % ncols]);
}
__global__ void mil_score_k(const float* __restrict__ T, const float* __restrict__ Wu,
                            const float* __restrict__ bu, float* __restrict__ araw, int n, int C) {
  int i = blockIdx.x * blockDim.x + threadIdx.x;
  if (i >= n) return;
  const float* tr = T + (size_t)i * C;
  float v = bu[0];
  for (int c = 0; c < C; ++c) v += tr[c] * Wu[c];
  araw[i] = v;
}

// ---------------- patient segment softmax + pool ----------------
__global__ void seg_max_k(const float* __restrict__ v, const int* __restrict__ pid,
                          unsigned* __restrict__ m, int n) {
  int i = blockIdx.x * blockDim.x + threadIdx.x;
  if (i < n) atomicMax(&m[pid[i]], f2ord(v[i]));
}
__global__ void seg_expsum_k(const float* __restrict__ v, const int* __restrict__ pid,
                             const unsigned* __restrict__ m, float* __restrict__ s,
                             float* __restrict__ ex, int n) {
  int i = blockIdx.x * blockDim.x + threadIdx.x;
  if (i >= n) return;
  int p = pid[i];
  float e = __expf(v[i] - ord2f(m[p]));
  ex[i] = e;
  atomicAdd(&s[p], e);
}
__global__ void seg_pool_k(float* __restrict__ ex_alpha, const int* __restrict__ pid,
                           const float* __restrict__ s, const float* __restrict__ h,
                           float* __restrict__ pool, int n, int C) {
  int i = blockIdx.x * blockDim.x + threadIdx.x;
  if (i >= n) return;
  int p = pid[i];
  float al = ex_alpha[i] / (s[p] + 1e-16f);
  ex_alpha[i] = al;                           // final alpha output
  const float* hr = h + (size_t)i * C;
  float* pr = pool + (size_t)p * C;
  for (int c = 0; c < C; ++c) atomicAdd(&pr[c], hr[c] * al);
}

// ---------------- classifier ----------------
__global__ void cls1_k(const float* __restrict__ hp, const float* __restrict__ W,
                       const float* __restrict__ b, float* __restrict__ t) {
  int i = blockIdx.x * blockDim.x + threadIdx.x;
  if (i >= PP * 64) return;
  int p = i >> 6, j = i & 63;
  const float* hr = hp + (size_t)p * HIDc;
  float v = b[j];
  for (int c = 0; c < HIDc; ++c) v += hr[c] * W[c * 64 + j];
  t[i] = (v > 0.f) ? v : 0.f;
}
__global__ void cls2_k(const float* __restrict__ t, const float* __restrict__ W,
                       const float* __restrict__ b, float* __restrict__ lg) {
  int i = blockIdx.x * blockDim.x + threadIdx.x;
  if (i >= PP * NCLS) return;
  int p = i / NCLS, k = i - p * NCLS;
  const float* tr = t + (size_t)p * 64;
  float v = b[k];
  for (int j = 0; j < 64; ++j) v += tr[j] * W[j * NCLS + k];
  lg[i] = v;
}

// ---------------- host launcher ----------------
extern "C" void kernel_launch(void* const* d_in, const int* in_sizes, int n_in,
                              void* d_out, int out_size, void* d_ws, size_t ws_size,
                              hipStream_t stream) {
  const float* x      = (const float*)d_in[0];
  const int*   ei     = (const int*)d_in[1];
  const int*   pid    = (const int*)d_in[2];
  const float* W1     = (const float*)d_in[4];
  const float* att_s1 = (const float*)d_in[5];
  const float* att_d1 = (const float*)d_in[6];
  const float* b1     = (const float*)d_in[7];
  const float* W2     = (const float*)d_in[8];
  const float* att_s2 = (const float*)d_in[9];
  const float* att_d2 = (const float*)d_in[10];
  const float* b2     = (const float*)d_in[11];
  const float* Wv     = (const float*)d_in[12];
  const float* bv     = (const float*)d_in[13];
  const float* Wu     = (const float*)d_in[14];
  const float* bu     = (const float*)d_in[15];
  const float* Wc1    = (const float*)d_in[16];
  const float* bc1    = (const float*)d_in[17];
  const float* Wc2    = (const float*)d_in[18];
  const float* bc2    = (const float*)d_in[19];

  // workspace carve
  char* w = (char*)d_ws;
  auto carve = [&](size_t bytes) -> char* {
    char* p = w; w += (bytes + 255) & ~(size_t)255; return p;
  };
  _Float16* Xh   = (_Float16*)carve((size_t)NN * IND * 2);
  _Float16* W1T  = (_Float16*)carve((size_t)F1 * IND * 2);
  float*    XL   = (float*)carve((size_t)NN * F1 * 4);
  float*    as1  = (float*)carve((size_t)NN * HDS * 4);
  float*    ad1  = (float*)carve((size_t)NN * HDS * 4);
  unsigned* m1   = (unsigned*)carve((size_t)NN * HDS * 4);
  float*    s1   = (float*)carve((size_t)NN * HDS * 4);
  float*    e1   = (float*)carve((size_t)ET * HDS * 4);
  float*    out1 = (float*)carve((size_t)NN * F1 * 4);
  _Float16* H1h  = (_Float16*)carve((size_t)NN * F1 * 2);
  _Float16* W2T  = (_Float16*)carve((size_t)HIDc * F1 * 2);
  float*    XL2  = (float*)carve((size_t)NN * HIDc * 4);
  float*    as2  = (float*)carve((size_t)NN * 4);
  float*    ad2  = (float*)carve((size_t)NN * 4);
  unsigned* m2   = (unsigned*)carve((size_t)NN * 4);
  float*    s2   = (float*)carve((size_t)NN * 4);
  float*    e2   = (float*)carve((size_t)ET * 4);
  float*    out2 = (float*)carve((size_t)NN * HIDc * 4);
  float*    h2f  = (float*)carve((size_t)NN * HIDc * 4);
  _Float16* h2h  = (_Float16*)carve((size_t)NN * HIDc * 2);
  _Float16* WvT  = (_Float16*)carve((size_t)ATTD * HIDc * 2);
  float*    Tac  = (float*)carve((size_t)NN * ATTD * 4);
  float*    araw = (float*)carve((size_t)NN * 4);
  unsigned* mp   = (unsigned*)carve((size_t)PP * 4);
  float*    sp   = (float*)carve((size_t)PP * 4);
  float*    ctmp = (float*)carve((size_t)PP * 64 * 4);

  // outputs: logits[500*7] | h_pool[500*128] | a[20000]
  float* logits = (float*)d_out;
  float* hpool  = logits + PP * NCLS;
  float* avec   = hpool + PP * HIDc;

  const int B = 256;

  // ---- conv1 ----
  cvt_f16_k<<<CDIV(NN * IND, B), B, 0, stream>>>(x, Xh, NN * IND);
  transpose_f16_k<<<CDIV(IND * F1, B), B, 0, stream>>>(W1, W1T, IND, F1);
  wmma_gemm_lds<<<CDIV(NN, 128) * (F1 / 64), B, 0, stream>>>(Xh, W1T, XL, NN, F1, IND);
  attn_dot_k<<<CDIV(NN * HDS, B), B, 0, stream>>>(XL, att_s1, as1, NN * HDS, HDS, HIDc);
  attn_dot_k<<<CDIV(NN * HDS, B), B, 0, stream>>>(XL, att_d1, ad1, NN * HDS, HDS, HIDc);
  fill_f32_k<<<CDIV(NN * HDS, B), B, 0, stream>>>((float*)m1, 0.0f, NN * HDS);  // ord floor
  fill_f32_k<<<CDIV(NN * HDS, B), B, 0, stream>>>(s1, 0.0f, NN * HDS);
  fill_f32_k<<<CDIV(NN * F1, B), B, 0, stream>>>(out1, 0.0f, NN * F1);
  edge_logits_k<<<CDIV(ET * HDS, B), B, 0, stream>>>(ei, as1, ad1, e1, m1, EE, ET, HDS);
  edge_expsum_k<<<CDIV(ET * HDS, B), B, 0, stream>>>(ei, m1, e1, s1, EE, ET, HDS);
  edge_aggr_k<<<CDIV(ET * HDS, B), B, 0, stream>>>(ei, XL, e1, s1, out1, EE, ET, HDS, HIDc);
  bias_elu_f16_k<<<CDIV(NN * F1, B), B, 0, stream>>>(out1, b1, H1h, NN * F1, F1);

  // ---- conv2 ----
  transpose_f16_k<<<CDIV(F1 * HIDc, B), B, 0, stream>>>(W2, W2T, F1, HIDc);
  wmma_gemm_lds<<<CDIV(NN, 128) * (HIDc / 64), B, 0, stream>>>(H1h, W2T, XL2, NN, HIDc, F1);
  attn_dot_k<<<CDIV(NN, B), B, 0, stream>>>(XL2, att_s2, as2, NN, 1, HIDc);
  attn_dot_k<<<CDIV(NN, B), B, 0, stream>>>(XL2, att_d2, ad2, NN, 1, HIDc);
  fill_f32_k<<<CDIV(NN, B), B, 0, stream>>>((float*)m2, 0.0f, NN);
  fill_f32_k<<<CDIV(NN, B), B, 0, stream>>>(s2, 0.0f, NN);
  fill_f32_k<<<CDIV(NN * HIDc, B), B, 0, stream>>>(out2, 0.0f, NN * HIDc);
  edge_logits_k<<<CDIV(ET, B), B, 0, stream>>>(ei, as2, ad2, e2, m2, EE, ET, 1);
  edge_expsum_k<<<CDIV(ET, B), B, 0, stream>>>(ei, m2, e2, s2, EE, ET, 1);
  edge_aggr_k<<<CDIV(ET, B), B, 0, stream>>>(ei, XL2, e2, s2, out2, EE, ET, 1, HIDc);
  bias_elu_both_k<<<CDIV(NN * HIDc, B), B, 0, stream>>>(out2, b2, h2f, h2h, NN * HIDc, HIDc);

  // ---- MIL gated attention pooling ----
  transpose_f16_k<<<CDIV(HIDc * ATTD, B), B, 0, stream>>>(Wv, WvT, HIDc, ATTD);
  wmma_gemm_lds<<<CDIV(NN, 128) * (ATTD / 64), B, 0, stream>>>(h2h, WvT, Tac, NN, ATTD, HIDc);
  tanh_bias_k<<<CDIV(NN * ATTD, B), B, 0, stream>>>(Tac, bv, NN * ATTD, ATTD);
  mil_score_k<<<CDIV(NN, B), B, 0, stream>>>(Tac, Wu, bu, araw, NN, ATTD);
  fill_f32_k<<<CDIV(PP, B), B, 0, stream>>>((float*)mp, 0.0f, PP);
  fill_f32_k<<<CDIV(PP, B), B, 0, stream>>>(sp, 0.0f, PP);
  fill_f32_k<<<CDIV(PP * HIDc, B), B, 0, stream>>>(hpool, 0.0f, PP * HIDc);
  seg_max_k<<<CDIV(NN, B), B, 0, stream>>>(araw, pid, mp, NN);
  seg_expsum_k<<<CDIV(NN, B), B, 0, stream>>>(araw, pid, mp, sp, avec, NN);
  seg_pool_k<<<CDIV(NN, B), B, 0, stream>>>(avec, pid, sp, h2f, hpool, NN, HIDc);

  // ---- classifier ----
  cls1_k<<<CDIV(PP * 64, B), B, 0, stream>>>(hpool, Wc1, bc1, ctmp);
  cls2_k<<<CDIV(PP * NCLS, B), B, 0, stream>>>(ctmp, Wc2, bc2, logits);
}
